// SelfAttention_46342697124278
// MI455X (gfx1250) — matrix operand order repaired and checked by reading, not compiled, not run on hardware
//
#include <hip/hip_runtime.h>
#include <hip/hip_bf16.h>
#include <stdint.h>

#define B_    2
#define T_    2048
#define HID_  1024
#define H_    16
#define D_    64
#define QKV_  3072
#define NTOK  (B_ * T_)

typedef __bf16 bf16_t;
typedef bf16_t v16bf __attribute__((ext_vector_type(16)));
typedef float  v8f   __attribute__((ext_vector_type(8)));
typedef unsigned short u16;
typedef u16 u16x8  __attribute__((ext_vector_type(8)));
typedef u16 u16x16 __attribute__((ext_vector_type(16)));
typedef unsigned int v4u __attribute__((ext_vector_type(4)));
typedef int v8i __attribute__((ext_vector_type(8)));
typedef int v4i __attribute__((ext_vector_type(4)));

#if defined(__has_builtin)
#if __has_builtin(__builtin_amdgcn_tensor_load_to_lds) && \
    __has_builtin(__builtin_amdgcn_s_wait_tensorcnt)
#define USE_TDM 1
#endif
#endif
#ifndef USE_TDM
#define USE_TDM 0
#endif

__device__ __forceinline__ u16 f2bf(float f) {
    union { float f; uint32_t u; } c; c.f = f;
    uint32_t u = c.u;
    u += 0x7FFFu + ((u >> 16) & 1u);           // round-to-nearest-even
    return (u16)(u >> 16);
}
__device__ __forceinline__ float bf2f(u16 s) {
    union { uint32_t u; float f; } c; c.u = ((uint32_t)s) << 16;
    return c.f;
}
// A layout used here (contiguous k, not the ISA table order): lane = m%16 | g<<4; elems 0..7 = K[16g..], 8..15 = K[16g+8..]
__device__ __forceinline__ v16bf mk_ab(u16x8 lo, u16x8 hi) {
    union { u16x16 v; u16x8 h[2]; } u;
    u.h[0] = lo; u.h[1] = hi;
    return __builtin_bit_cast(v16bf, u.v);
}
// Contiguous 32B fragment (B layout: elem j = B[k=j+16g][n=lane%16])
__device__ __forceinline__ v16bf ld16(const u16* p) {
    return __builtin_bit_cast(v16bf, *(const u16x16*)p);
}
__device__ __forceinline__ v8f wmma_bf16(v16bf a, v16bf b, v8f c) {
    return __builtin_amdgcn_wmma_f32_16x16x32_bf16(false, a, false, b,
                                                   (short)0, c, false, false);
}

#if USE_TDM
// Issue a 2D TDM tile load (global row-major -> LDS, packed rows).
// Units are 8-byte (data_size code 3). One issue per executing wave.
__device__ __forceinline__ void tdm_load_2d(uint32_t lds_off, uint64_t gaddr,
                                            uint32_t tensor_d0_u,  // row len (units)
                                            uint32_t tensor_d1,    // #rows in tensor
                                            uint32_t tile_d0_u,    // tile row len (units)
                                            uint32_t tile_d1,      // tile #rows
                                            uint64_t stride0_u) {  // row stride (units)
    v4u g0;
    g0[0] = 1u;                                         // count=1, user mode
    g0[1] = lds_off;                                    // lds_addr
    g0[2] = (uint32_t)(gaddr & 0xFFFFFFFFu);            // global_addr[31:0]
    g0[3] = (uint32_t)((gaddr >> 32) & 0x01FFFFFFu)     // global_addr[56:32]
          | 0x80000000u;                                // type=2 ("image")
    v8i g1;
    g1[0] = (int)(3u << 16);                            // wg_mask=0, data_size=8B
    g1[1] = (int)((tensor_d0_u & 0xFFFFu) << 16);
    g1[2] = (int)(((tensor_d0_u >> 16) & 0xFFFFu) | ((tensor_d1 & 0xFFFFu) << 16));
    g1[3] = (int)(((tensor_d1 >> 16) & 0xFFFFu) | ((tile_d0_u & 0xFFFFu) << 16));
    g1[4] = (int)(tile_d1 & 0xFFFFu);                   // tile_dim1; tile_dim2=0
    g1[5] = (int)(uint32_t)(stride0_u & 0xFFFFFFFFu);   // dim0 stride lo
    g1[6] = (int)(uint32_t)((stride0_u >> 32) & 0xFFFFu); // dim0 stride hi; dim1 stride=0
    g1[7] = 0;
    v4i z4 = {0, 0, 0, 0};
    v8i z8 = {0, 0, 0, 0, 0, 0, 0, 0};
    __builtin_amdgcn_tensor_load_to_lds(g0, g1, z4, z4, z8, 0);
}
__device__ __forceinline__ uint32_t lds_off_of(const void* p) {
    return (uint32_t)(uintptr_t)p;   // generic LDS pointer: low 32 bits = LDS offset
}
#endif

// ---------------------------------------------------------------- cvt f32->bf16
__global__ __launch_bounds__(256) void cvt_f32_bf16(const float* __restrict__ src,
                                                    u16* __restrict__ dst, int n) {
    int i = blockIdx.x * blockDim.x + threadIdx.x;
    int stride = gridDim.x * blockDim.x;
    for (; i < n; i += stride) dst[i] = f2bf(src[i]);
}

// ====================================================================== GEMMs
// Block tile 128(M) x 64(N), BK=32, 4 waves; wave owns 32x64 (2x4 WMMA frags).
// TDM path: wave0 DMA-loads both tiles, double-buffered behind WMMA.

#define GEMM_BODY()                                                             \
    const int tid = threadIdx.x;                                                \
    const int wave = tid >> 5, lane = tid & 31;                                 \
    const int nl = lane & 15, g = lane >> 4;                                    \
    const int m0 = blockIdx.y * 128;                                            \
    const int n0 = blockIdx.x * 64;                                             \
    v8f acc[2][4] = {};

__device__ __forceinline__ void gemm_compute_step(const u16 (*Xs)[32],
                                                  const u16 (*Ws)[32],
                                                  int wave, int nl, int g,
                                                  v8f acc[2][4]) {
    v16bf a[2];
    #pragma unroll
    for (int mt = 0; mt < 2; ++mt) {
        const int ar = wave * 32 + mt * 16 + nl;
        a[mt] = mk_ab(*(const u16x8*)&Xs[ar][16 * g],
                      *(const u16x8*)&Xs[ar][16 * g + 8]);
    }
    #pragma unroll
    for (int nt = 0; nt < 4; ++nt) {
        v16bf b = ld16(&Ws[nt * 16 + nl][16 * g]);
        #pragma unroll
        for (int mt = 0; mt < 2; ++mt)
            acc[mt][nt] = wmma_bf16(a[mt], b, acc[mt][nt]);
    }
}

__global__ __launch_bounds__(128) void qkv_gemm(const u16* __restrict__ xb,
                                                const u16* __restrict__ wb,
                                                const float* __restrict__ bias,
                                                u16* __restrict__ qo,
                                                u16* __restrict__ ko,
                                                u16* __restrict__ vo) {
    __shared__ __align__(32) u16 Xs[2][128][32];
    __shared__ __align__(32) u16 Ws[2][64][32];
    GEMM_BODY()
#if USE_TDM
    if (wave == 0) {
        tdm_load_2d(lds_off_of(&Xs[0][0][0]), (uint64_t)(uintptr_t)&xb[(size_t)m0 * HID_],
                    HID_ / 4, NTOK, 8, 128, HID_ / 4);
        tdm_load_2d(lds_off_of(&Ws[0][0][0]), (uint64_t)(uintptr_t)&wb[(size_t)n0 * HID_],
                    HID_ / 4, QKV_, 8, 64, HID_ / 4);
        __builtin_amdgcn_s_wait_tensorcnt(0);
    }
    __syncthreads();
    int cur = 0;
    for (int kk = 0; kk < HID_; kk += 32) {
        const int nxt = cur ^ 1;
        if (kk + 32 < HID_ && wave == 0) {
            tdm_load_2d(lds_off_of(&Xs[nxt][0][0]),
                        (uint64_t)(uintptr_t)&xb[(size_t)m0 * HID_ + kk + 32],
                        HID_ / 4, NTOK, 8, 128, HID_ / 4);
            tdm_load_2d(lds_off_of(&Ws[nxt][0][0]),
                        (uint64_t)(uintptr_t)&wb[(size_t)n0 * HID_ + kk + 32],
                        HID_ / 4, QKV_, 8, 64, HID_ / 4);
        }
        gemm_compute_step(Xs[cur], Ws[cur], wave, nl, g, acc);
        if (wave == 0) __builtin_amdgcn_s_wait_tensorcnt(0);
        __syncthreads();
        cur = nxt;
    }
#else
    for (int kk = 0; kk < HID_; kk += 32) {
        #pragma unroll
        for (int p = 0; p < 4; ++p) {
            int lin = p * 1024 + tid * 8;
            int r = lin >> 5, c = lin & 31;
            *(u16x8*)&Xs[0][r][c] = *(const u16x8*)&xb[(size_t)(m0 + r) * HID_ + kk + c];
        }
        #pragma unroll
        for (int p = 0; p < 2; ++p) {
            int lin = p * 1024 + tid * 8;
            int r = lin >> 5, c = lin & 31;
            *(u16x8*)&Ws[0][r][c] = *(const u16x8*)&wb[(size_t)(n0 + r) * HID_ + kk + c];
        }
        __syncthreads();
        gemm_compute_step(Xs[0], Ws[0], wave, nl, g, acc);
        __syncthreads();
    }
#endif
    const int s = n0 / 1024;                 // 0=q, 1=k, 2=v
    const int h = (n0 % 1024) / 64;
    u16* dst = (s == 0) ? qo : (s == 1) ? ko : vo;
    #pragma unroll
    for (int nt = 0; nt < 4; ++nt) {
        const int d  = nt * 16 + nl;
        const float bv = bias[n0 + d];
        #pragma unroll
        for (int mt = 0; mt < 2; ++mt)
            #pragma unroll
            for (int vv = 0; vv < 8; ++vv) {
                int tok = m0 + wave * 32 + mt * 16 + vv + 8 * g;
                int b_ = tok >> 11, t = tok & (T_ - 1);
                dst[(((size_t)(b_ * H_ + h) * T_) + t) * D_ + d] =
                    f2bf(acc[mt][nt][vv] + bv);
            }
    }
}

__global__ __launch_bounds__(128) void out_gemm(const u16* __restrict__ ab,
                                                const u16* __restrict__ wb,
                                                const float* __restrict__ bias,
                                                float* __restrict__ out) {
    __shared__ __align__(32) u16 Xs[2][128][32];
    __shared__ __align__(32) u16 Ws[2][64][32];
    GEMM_BODY()
#if USE_TDM
    if (wave == 0) {
        tdm_load_2d(lds_off_of(&Xs[0][0][0]), (uint64_t)(uintptr_t)&ab[(size_t)m0 * HID_],
                    HID_ / 4, NTOK, 8, 128, HID_ / 4);
        tdm_load_2d(lds_off_of(&Ws[0][0][0]), (uint64_t)(uintptr_t)&wb[(size_t)n0 * HID_],
                    HID_ / 4, HID_, 8, 64, HID_ / 4);
        __builtin_amdgcn_s_wait_tensorcnt(0);
    }
    __syncthreads();
    int cur = 0;
    for (int kk = 0; kk < HID_; kk += 32) {
        const int nxt = cur ^ 1;
        if (kk + 32 < HID_ && wave == 0) {
            tdm_load_2d(lds_off_of(&Xs[nxt][0][0]),
                        (uint64_t)(uintptr_t)&ab[(size_t)m0 * HID_ + kk + 32],
                        HID_ / 4, NTOK, 8, 128, HID_ / 4);
            tdm_load_2d(lds_off_of(&Ws[nxt][0][0]),
                        (uint64_t)(uintptr_t)&wb[(size_t)n0 * HID_ + kk + 32],
                        HID_ / 4, HID_, 8, 64, HID_ / 4);
        }
        gemm_compute_step(Xs[cur], Ws[cur], wave, nl, g, acc);
        if (wave == 0) __builtin_amdgcn_s_wait_tensorcnt(0);
        __syncthreads();
        cur = nxt;
    }
#else
    for (int kk = 0; kk < HID_; kk += 32) {
        #pragma unroll
        for (int p = 0; p < 4; ++p) {
            int lin = p * 1024 + tid * 8;
            int r = lin >> 5, c = lin & 31;
            *(u16x8*)&Xs[0][r][c] = *(const u16x8*)&ab[(size_t)(m0 + r) * HID_ + kk + c];
        }
        #pragma unroll
        for (int p = 0; p < 2; ++p) {
            int lin = p * 1024 + tid * 8;
            int r = lin >> 5, c = lin & 31;
            *(u16x8*)&Ws[0][r][c] = *(const u16x8*)&wb[(size_t)(n0 + r) * HID_ + kk + c];
        }
        __syncthreads();
        gemm_compute_step(Xs[0], Ws[0], wave, nl, g, acc);
        __syncthreads();
    }
#endif
    #pragma unroll
    for (int nt = 0; nt < 4; ++nt) {
        const int d = nt * 16 + nl;
        const float bv = bias[n0 + d];
        #pragma unroll
        for (int mt = 0; mt < 2; ++mt)
            #pragma unroll
            for (int vv = 0; vv < 8; ++vv) {
                size_t row = (size_t)(m0 + wave * 32 + mt * 16 + vv + 8 * g);
                out[row * HID_ + n0 + d] = acc[mt][nt][vv] + bv;
            }
    }
}

// ---------------------------------------------------------------- RoPE in place
__global__ __launch_bounds__(256) void rope_kernel(u16* __restrict__ q,
                                                   u16* __restrict__ k) {
    int tid = blockIdx.x * blockDim.x + threadIdx.x;
    if (tid >= B_ * H_ * T_ * 16) return;
    int i  = tid & 15;
    int t  = (tid >> 4) & (T_ - 1);
    int bh = tid >> 15;
    size_t base = ((size_t)bh * T_ + t) * D_ + 2 * i;
    float inv = __expf(-((2.0f * (float)i) / 32.0f) * 9.210340371976184f); // ln(1e4)
    float ang = (float)t * inv;
    float c = __cosf(ang), s = __sinf(ang);
    {
        float x1 = bf2f(q[base]), x2 = bf2f(q[base + 1]);
        q[base]     = f2bf(x1 * c - x2 * s);
        q[base + 1] = f2bf(x1 * s + x2 * c);
    }
    {
        float x1 = bf2f(k[base]), x2 = bf2f(k[base + 1]);
        k[base]     = f2bf(x1 * c - x2 * s);
        k[base + 1] = f2bf(x1 * s + x2 * c);
    }
}

// ------------------------------------------------------------- causal flash attn
// Grid (T/64, B*H); 4 waves; wave owns 16 query rows. K tile via TDM, V manual
// (transposed into LDS, overlapped with the in-flight K DMA).
__global__ __launch_bounds__(128) void flash_attn(const u16* __restrict__ q,
                                                  const u16* __restrict__ k,
                                                  const u16* __restrict__ v,
                                                  u16* __restrict__ o) {
    __shared__ __align__(32) u16 Ks[64][64];
    __shared__ __align__(32) u16 VTs[64][64];   // V transposed: [d][keypos]
    __shared__ __align__(32) u16 Sh[64][64];    // Q staging, then per-wave P
    const int tid = threadIdx.x;
    const int wave = tid >> 5, lane = tid & 31;
    const int nl = lane & 15, g = lane >> 4;
    const int qb0 = blockIdx.x * 64;
    const int bh  = blockIdx.y;
    const size_t plane = (size_t)bh * T_ * D_;

#if USE_TDM
    if (wave == 0) {
        tdm_load_2d(lds_off_of(&Sh[0][0]),
                    (uint64_t)(uintptr_t)&q[plane + (size_t)qb0 * D_],
                    D_ / 4, T_, D_ / 4, 64, D_ / 4);
        __builtin_amdgcn_s_wait_tensorcnt(0);
    }
#else
    #pragma unroll
    for (int p = 0; p < 8; ++p) {
        int lin = p * 1024 + tid * 8;
        int r = lin >> 6, c = lin & 63;
        *(u16x8*)&Sh[r][c] = *(const u16x8*)&q[plane + (size_t)(qb0 + r) * D_ + c];
    }
#endif
    __syncthreads();
    v16bf aQ[2];
    {
        const int m = wave * 16 + nl;
        #pragma unroll
        for (int f = 0; f < 2; ++f)
            aQ[f] = mk_ab(*(const u16x8*)&Sh[m][32 * f + 16 * g],
                          *(const u16x8*)&Sh[m][32 * f + 16 * g + 8]);
    }
    float mi[8], li[8];
    v8f accO[4] = {};
    #pragma unroll
    for (int vv = 0; vv < 8; ++vv) { mi[vv] = -3.0e38f; li[vv] = 0.0f; }

    const int nkb = blockIdx.x + 1;
    for (int kbi = 0; kbi < nkb; ++kbi) {
        const int kb0 = kbi * 64;
        __syncthreads();
#if USE_TDM
        if (wave == 0)
            tdm_load_2d(lds_off_of(&Ks[0][0]),
                        (uint64_t)(uintptr_t)&k[plane + (size_t)kb0 * D_],
                        D_ / 4, T_, D_ / 4, 64, D_ / 4);
#endif
        #pragma unroll
        for (int p = 0; p < 8; ++p) {
            int lin = p * 1024 + tid * 8;
            int r = lin >> 6, c = lin & 63;
#if !USE_TDM
            *(u16x8*)&Ks[r][c] = *(const u16x8*)&k[plane + (size_t)(kb0 + r) * D_ + c];
#endif
            u16x8 vv8 = *(const u16x8*)&v[plane + (size_t)(kb0 + r) * D_ + c];
            #pragma unroll
            for (int j = 0; j < 8; ++j) VTs[c + j][r] = vv8[j];
        }
        if (kbi + 1 < nkb) {   // prefetch next V tile (global_prefetch_b8)
            const u16* pv = &v[plane + (size_t)(kb0 + 64) * D_];
            __builtin_prefetch(pv + (size_t)tid * 64, 0, 0);
        }
#if USE_TDM
        if (wave == 0) __builtin_amdgcn_s_wait_tensorcnt(0);
#endif
        __syncthreads();

        v8f S[4] = {};
        #pragma unroll
        for (int nt = 0; nt < 4; ++nt)
            #pragma unroll
            for (int f = 0; f < 2; ++f)
                S[nt] = wmma_bf16(aQ[f], ld16(&Ks[nt * 16 + nl][32 * f + 16 * g]), S[nt]);

        const bool last = (kbi == nkb - 1);
        float rmax[8];
        #pragma unroll
        for (int vv = 0; vv < 8; ++vv) rmax[vv] = -3.0e38f;
        #pragma unroll
        for (int nt = 0; nt < 4; ++nt)
            #pragma unroll
            for (int vv = 0; vv < 8; ++vv) {
                float sc = S[nt][vv] * 0.125f;                     // 1/sqrt(64)
                if (last && (kb0 + nt * 16 + nl) > (qb0 + wave * 16 + vv + 8 * g))
                    sc = -1.0e9f;                                  // MASK_VAL
                S[nt][vv] = sc;
                rmax[vv] = fmaxf(rmax[vv], sc);
            }
        #pragma unroll
        for (int vv = 0; vv < 8; ++vv) {
            float r = rmax[vv];
            r = fmaxf(r, __shfl_xor(r, 1, 32));
            r = fmaxf(r, __shfl_xor(r, 2, 32));
            r = fmaxf(r, __shfl_xor(r, 4, 32));
            r = fmaxf(r, __shfl_xor(r, 8, 32));
            float mnew  = fmaxf(mi[vv], r);
            float alpha = __expf(mi[vv] - mnew);
            mi[vv] = mnew;
            li[vv] *= alpha;
            #pragma unroll
            for (int nt = 0; nt < 4; ++nt) accO[nt][vv] *= alpha;
            float rs = 0.0f;
            #pragma unroll
            for (int nt = 0; nt < 4; ++nt) {
                float p = __expf(S[nt][vv] - mnew);
                S[nt][vv] = p;
                rs += p;
            }
            rs += __shfl_xor(rs, 1, 32);
            rs += __shfl_xor(rs, 2, 32);
            rs += __shfl_xor(rs, 4, 32);
            rs += __shfl_xor(rs, 8, 32);
            li[vv] += rs;
        }
        // P (C layout) -> wave-private LDS slice -> reload in A layout
        #pragma unroll
        for (int nt = 0; nt < 4; ++nt)
            #pragma unroll
            for (int vv = 0; vv < 8; ++vv)
                Sh[wave * 16 + vv + 8 * g][nt * 16 + nl] = f2bf(S[nt][vv]);
        #pragma unroll
        for (int f = 0; f < 2; ++f) {
            const int m = wave * 16 + nl;
            v16bf aP = mk_ab(*(const u16x8*)&Sh[m][32 * f + 16 * g],
                             *(const u16x8*)&Sh[m][32 * f + 16 * g + 8]);
            #pragma unroll
            for (int nt = 0; nt < 4; ++nt)
                accO[nt] = wmma_bf16(aP, ld16(&VTs[nt * 16 + nl][32 * f + 16 * g]),
                                     accO[nt]);
        }
        __syncthreads();
    }
    const int b_ = bh >> 4, h = bh & 15;
    #pragma unroll
    for (int nt = 0; nt < 4; ++nt) {
        const int d = nt * 16 + nl;
        #pragma unroll
        for (int vv = 0; vv < 8; ++vv) {
            int trow = qb0 + wave * 16 + vv + 8 * g;
            size_t tok = (size_t)b_ * T_ + trow;
            o[tok * (H_ * D_) + h * D_ + d] = f2bf(accO[nt][vv] / li[vv]);
        }
    }
}

// -------------------------------------------------------------------- launcher
extern "C" void kernel_launch(void* const* d_in, const int* in_sizes, int n_in,
                              void* d_out, int out_size, void* d_ws, size_t ws_size,
                              hipStream_t stream) {
    const float* x      = (const float*)d_in[0];
    // d_in[1] = mask (bool) -- statically causal, unused
    const float* qkv_w  = (const float*)d_in[2];
    const float* qkv_b  = (const float*)d_in[3];
    const float* out_w  = (const float*)d_in[4];
    const float* out_b  = (const float*)d_in[5];
    float* out          = (float*)d_out;

    char* w = (char*)d_ws;
    u16* xb    = (u16*)(w);                               // 8 MB
    u16* wqkvb = (u16*)(w + (8u  << 20));                 // 6 MB
    u16* woutb = (u16*)(w + (14u << 20) + (512u << 10));  // 2 MB
    u16* qb    = (u16*)(w + (16u << 20) + (512u << 10));  // 8 MB
    u16* kb    = (u16*)(w + (24u << 20) + (512u << 10));  // 8 MB
    u16* vb    = (u16*)(w + (32u << 20) + (512u << 10));  // 8 MB
    u16* ab    = (u16*)(w + (40u << 20) + (512u << 10));  // 8 MB

    cvt_f32_bf16<<<1024, 256, 0, stream>>>(x,     xb,    NTOK * HID_);
    cvt_f32_bf16<<<1024, 256, 0, stream>>>(qkv_w, wqkvb, QKV_ * HID_);
    cvt_f32_bf16<<<1024, 256, 0, stream>>>(out_w, woutb, HID_ * HID_);

    qkv_gemm<<<dim3(QKV_ / 64, NTOK / 128), 128, 0, stream>>>(xb, wqkvb, qkv_b,
                                                              qb, kb, vb);
    rope_kernel<<<(B_ * H_ * T_ * 16) / 256, 256, 0, stream>>>(qb, kb);
    flash_attn<<<dim3(T_ / 64, B_ * H_), 128, 0, stream>>>(qb, kb, vb, ab);
    out_gemm<<<dim3(HID_ / 64, NTOK / 128), 128, 0, stream>>>(ab, woutb, out_b, out);
}
